// dynamic_graph_constructor_5918464934354
// MI455X (gfx1250) — matrix operand order repaired and checked by reading, not compiled, not run on hardware
//
#include <hip/hip_runtime.h>
#include <hip/hip_bf16.h>

typedef __attribute__((ext_vector_type(16))) _Float16 v16h;
typedef __attribute__((ext_vector_type(8)))  float    v8f;

#define BB 16
#define NN 2048
#define DD 64
#define KSEL 20

#define LOG2E 1.4426950408889634f

// Branchless fast transcendentals on v_exp_f32 / v_rcp_f32 (no EXEC branches).
static __device__ inline float fast_sigmoid(float x) {
    return __builtin_amdgcn_rcpf(1.0f + __builtin_exp2f(-LOG2E * x));
}
static __device__ inline float fast_tanh3(float x) {      // tanh(3x), full range
    float e = __builtin_exp2f((6.0f * LOG2E) * x);        // exp(6x)
    return 1.0f - 2.0f * __builtin_amdgcn_rcpf(e + 1.0f); // x->-inf: e->0 -> -1
}
static __device__ inline float relu_tanh3(float a) {      // relu(tanh(3a))
    float e = __builtin_exp2f((6.0f * LOG2E) * a);
    float t = 1.0f - 2.0f * __builtin_amdgcn_rcpf(e + 1.0f);
    return a > 0.0f ? t : 0.0f;                           // v_cndmask, no branch
}

// ---------------------------------------------------------------------------
// Stage 1: per-(b,n) gated row transform -> n1, n2 (f16) in workspace.
// One 64-thread block per row; thread t owns output feature t of each linear.
// ---------------------------------------------------------------------------
__global__ __launch_bounds__(64)
void stage1_rows(const int* __restrict__ idx,
                 const float* __restrict__ emb,
                 const float* __restrict__ W1, const float* __restrict__ b1,
                 const float* __restrict__ W2, const float* __restrict__ b2,
                 const float* __restrict__ Wgs, const float* __restrict__ bgs,
                 const float* __restrict__ Wgd, const float* __restrict__ bgd,
                 const float* __restrict__ emb1, const float* __restrict__ emb2,
                 _Float16* __restrict__ n1h, _Float16* __restrict__ n2h)
{
    __shared__ float e[DD], s1[DD], s2[DD], h1[DD], h2[DD];
    const int bid = blockIdx.x;          // 0 .. B*N-1
    const int b = bid / NN;
    const int n = bid % NN;
    const int t = threadIdx.x;           // 0..63

    const int node = idx[n];
    const size_t rowOff = ((size_t)b * NN + n) * DD;
    e[t]  = emb[rowOff + t];
    s1[t] = emb1[(size_t)node * DD + t];
    s2[t] = emb2[(size_t)node * DD + t];
    __syncthreads();

    float gd = bgd[t];
    float a1 = bgs[t], a2 = a1;
    const float* wgd = Wgd + t * DD;
    const float* wgs = Wgs + t * DD;
    #pragma unroll 8
    for (int k = 0; k < DD; ++k) {
        gd += wgd[k] * e[k];
        float w = wgs[k];
        a1 += w * s1[k];
        a2 += w * s2[k];
    }
    float g1 = fast_sigmoid(a1 + gd);
    float g2 = fast_sigmoid(a2 + gd);
    h1[t] = (1.0f - g1) * e[t] + g1 * s1[t];
    h2[t] = (1.0f - g2) * e[t] + g2 * s2[t];
    __syncthreads();

    float o1 = b1[t], o2 = b2[t];
    const float* w1 = W1 + t * DD;
    const float* w2 = W2 + t * DD;
    #pragma unroll 8
    for (int k = 0; k < DD; ++k) {
        o1 += w1[k] * h1[k];
        o2 += w2[k] * h2[k];
    }
    n1h[rowOff + t] = (_Float16)fast_tanh3(o1);
    n2h[rowOff + t] = (_Float16)fast_tanh3(o2);
}

// ---------------------------------------------------------------------------
// Stage 2: fused antisymmetric WMMA GEMM + relu(tanh) + noise + top-K + write.
// Block = 256 threads (8 waves) owns (batch b, 16 rows). Dynamic LDS holds the
// 16 x 2048 f32 score strip (128 KB -- CDNA5 WGP LDS is 320 KB).
// ---------------------------------------------------------------------------
static __device__ inline v8f wmma_f16(v16h a, v16h bm, v8f c) {
    return __builtin_amdgcn_wmma_f32_16x16x32_f16(
        false, a, false, bm, (short)0, c, false, false);
}

__global__ __launch_bounds__(256)
void stage2_adj_topk(const _Float16* __restrict__ n1h,
                     const _Float16* __restrict__ n2h,
                     const float* __restrict__ noise,
                     float* __restrict__ out)
{
    extern __shared__ float score[];     // [16][NN]
    const int bid  = blockIdx.x;         // 0 .. B*(N/16)-1
    const int b    = bid >> 7;           // / (N/16)
    const int n0   = (bid & 127) << 4;   // row-tile base
    const int tid  = threadIdx.x;
    const int wave = tid >> 5;           // 0..7
    const int lane = tid & 31;
    const int lm   = lane & 15;
    const int lhi  = lane >> 4;          // 0 or 1

    const _Float16* n1b = n1h + (size_t)b * NN * DD;
    const _Float16* n2b = n2h + (size_t)b * NN * DD;

    // --- A operands: rows n0..n0+15 of n1 and n2, K split into [0,32) / [32,64)
    // A 16x32 in contiguous K order (same as B): lane holds M=lm; halves 0..7 -> K = kb+0..7,
    // halves 8..15 -> K = kb+8..15, kb = lhi*16.
    v16h a1lo, a1hi, a2lo, a2hi;
    {
        const _Float16* r1 = n1b + (size_t)(n0 + lm) * DD;
        const _Float16* r2 = n2b + (size_t)(n0 + lm) * DD;
        const int kb = lhi * 16;
        #pragma unroll
        for (int j = 0; j < 8; ++j) {
            a1lo[j]     = r1[kb + j];
            a1lo[j + 8] = r1[kb + 8 + j];
            a1hi[j]     = r1[32 + kb + j];
            a1hi[j + 8] = r1[32 + kb + 8 + j];
            a2lo[j]     = r2[kb + j];
            a2lo[j + 8] = r2[kb + 8 + j];
            a2hi[j]     = r2[32 + kb + j];
            a2hi[j + 8] = r2[32 + kb + 8 + j];
        }
    }

    // --- GEMM phase: wave owns 16 column tiles of 16.
    // B 32x16 layout: lane holds column N=lm, 16 contiguous K halves starting at
    // lhi*16 -> one 32-byte load (2x b128).
    for (int t = 0; t < 16; ++t) {
        const int m0 = (wave * 16 + t) * 16;
        const _Float16* c2 = n2b + (size_t)(m0 + lm) * DD + lhi * 16;
        const _Float16* c1 = n1b + (size_t)(m0 + lm) * DD + lhi * 16;
        v16h b1lo = *(const v16h*)(c2);
        v16h b1hi = *(const v16h*)(c2 + 32);
        v16h b2lo = *(const v16h*)(c1);
        v16h b2hi = *(const v16h*)(c1 + 32);

        v8f acc1 = {};
        v8f acc2 = {};
        acc1 = wmma_f16(a1lo, b1lo, acc1);   // n1[n] . n2[m]
        acc1 = wmma_f16(a1hi, b1hi, acc1);
        acc2 = wmma_f16(a2lo, b2lo, acc2);   // n2[n] . n1[m]
        acc2 = wmma_f16(a2hi, b2hi, acc2);

        // C layout: c[i] -> row = i + lhi*8, col = lm (within tile)
        #pragma unroll
        for (int i = 0; i < 8; ++i) {
            const int row = i + lhi * 8;
            const int col = m0 + lm;
            const float a = acc1[i] - acc2[i];
            const float adj = relu_tanh3(a);
            const float nz = __builtin_nontemporal_load(
                &noise[((size_t)b * NN + (n0 + row)) * NN + col]);
            score[row * NN + col] = adj + 0.01f * nz;
        }
    }
    __syncthreads();

    // --- Top-K phase: wave owns rows 2*wave and 2*wave+1.
    // Winners encoded in-place as -(adj+1) (valid scores are >= 0).
    for (int rr = 0; rr < 2; ++rr) {
        const int row = wave * 2 + rr;
        float* srow = score + row * NN;
        const size_t nrow = ((size_t)b * NN + (n0 + row)) * (size_t)NN;
        for (int k = 0; k < KSEL; ++k) {
            float best = -0.5f;
            int   bj   = lane;
            for (int j = lane; j < NN; j += 32) {
                float v = srow[j];
                if (v > best) { best = v; bj = j; }
            }
            #pragma unroll
            for (int off = 16; off > 0; off >>= 1) {
                float ov = __shfl_xor(best, off, 32);
                int   oj = __shfl_xor(bj, off, 32);
                if (ov > best || (ov == best && oj < bj)) { best = ov; bj = oj; }
            }
            if (lane == 0) {
                const float adj = best - 0.01f * noise[nrow + bj];
                srow[bj] = -(adj + 1.0f);
            }
            __syncthreads();   // uniform across all waves (2*KSEL rounds each)
        }
    }

    // --- Output: masked row write, zero elsewhere; non-temporal (stream once).
    for (int rr = 0; rr < 2; ++rr) {
        const int row = wave * 2 + rr;
        const float* srow = score + row * NN;
        float* orow = out + ((size_t)b * NN + (n0 + row)) * (size_t)NN;
        for (int j = lane; j < NN; j += 32) {
            const float v = srow[j];
            const float o = (v < -0.5f) ? (-v - 1.0f) : 0.0f;
            __builtin_nontemporal_store(o, &orow[j]);
        }
    }
}

// ---------------------------------------------------------------------------
extern "C" void kernel_launch(void* const* d_in, const int* in_sizes, int n_in,
                              void* d_out, int out_size, void* d_ws, size_t ws_size,
                              hipStream_t stream) {
    const int*   idx   = (const int*)  d_in[0];
    const float* emb   = (const float*)d_in[1];
    const float* noise = (const float*)d_in[2];
    const float* W1    = (const float*)d_in[3];
    const float* b1    = (const float*)d_in[4];
    const float* W2    = (const float*)d_in[5];
    const float* b2    = (const float*)d_in[6];
    const float* Wgs   = (const float*)d_in[7];
    const float* bgs   = (const float*)d_in[8];
    const float* Wgd   = (const float*)d_in[9];
    const float* bgd   = (const float*)d_in[10];
    const float* emb1  = (const float*)d_in[11];
    const float* emb2  = (const float*)d_in[12];
    float* out = (float*)d_out;

    _Float16* n1h = (_Float16*)d_ws;
    _Float16* n2h = n1h + (size_t)BB * NN * DD;

    // Stage 1: one 64-thread block per (b, n) row.
    stage1_rows<<<BB * NN, DD, 0, stream>>>(idx, emb, W1, b1, W2, b2,
                                            Wgs, bgs, Wgd, bgd, emb1, emb2,
                                            n1h, n2h);

    // Stage 2: one 256-thread block per (b, 16-row tile); 128 KB dynamic LDS.
    const int blocks = BB * (NN / 16);
    const size_t lds = (size_t)16 * NN * sizeof(float);
    stage2_adj_topk<<<blocks, 256, lds, stream>>>(n1h, n2h, noise, out);
}
